// SegmentConv2d_18811956756585
// MI455X (gfx1250) — 8-bit matrix operand order repaired and checked by reading, compiled, not run on hardware
//
#include <hip/hip_runtime.h>

typedef __attribute__((ext_vector_type(8))) int          v8i;
typedef __attribute__((ext_vector_type(2))) unsigned int v2u;
typedef __attribute__((ext_vector_type(4))) unsigned int v4u;
typedef __attribute__((ext_vector_type(4))) float        v4f;

static constexpr int   Nn = 8, Cc = 256, Hh = 128, Ww = 128, OC = 512, COC = 128;
static constexpr float QMAX = 127.0f;
static constexpr size_t X_ELEMS = (size_t)Nn * Cc * Hh * Ww;
static constexpr size_t W_ELEMS = (size_t)OC * Cc * 9;
static constexpr size_t WS_XQ_OFF = 256;
static constexpr size_t WS_WQ_OFF = WS_XQ_OFF + X_ELEMS;

#define USE_ASYNC_LDS 1

__global__ void zero_slots_kernel(unsigned int* slots) {
    if (threadIdx.x < 8) slots[threadIdx.x] = 0u;
}

__global__ __launch_bounds__(256) void amax_x_kernel(const float* __restrict__ x,
                                                     unsigned int* __restrict__ slots) {
    __shared__ unsigned int red[256];
    unsigned int m = 0u;
    size_t stride = (size_t)gridDim.x * 256;
    for (size_t i = (size_t)blockIdx.x * 256 + threadIdx.x; i < X_ELEMS; i += stride)
        m = max(m, __float_as_uint(fabsf(x[i])));
    red[threadIdx.x] = m;
    __syncthreads();
    for (int s = 128; s > 0; s >>= 1) {
        if (threadIdx.x < (unsigned)s) red[threadIdx.x] = max(red[threadIdx.x], red[threadIdx.x + s]);
        __syncthreads();
    }
    if (threadIdx.x == 0) atomicMax(&slots[0], red[0]);
}

__global__ __launch_bounds__(256) void amax_w_kernel(const float* __restrict__ w,
                                                     unsigned int* __restrict__ slots) {
    __shared__ unsigned int red[256];
    const size_t CE = (size_t)COC * Cc * 9;
    int chunk = blockIdx.x >> 6;
    int b     = blockIdx.x & 63;
    const float* base = w + (size_t)chunk * CE;
    unsigned int m = 0u;
    for (size_t i = (size_t)b * 256 + threadIdx.x; i < CE; i += (size_t)64 * 256)
        m = max(m, __float_as_uint(fabsf(base[i])));
    red[threadIdx.x] = m;
    __syncthreads();
    for (int s = 128; s > 0; s >>= 1) {
        if (threadIdx.x < (unsigned)s) red[threadIdx.x] = max(red[threadIdx.x], red[threadIdx.x + s]);
        __syncthreads();
    }
    if (threadIdx.x == 0) atomicMax(&slots[1 + chunk], red[0]);
}

__global__ __launch_bounds__(256) void quant_x_kernel(const float* __restrict__ x,
                                                      const unsigned int* __restrict__ slots,
                                                      char* __restrict__ xq) {
    __shared__ char q[64][256];
    int bid  = blockIdx.x;
    int wseg = bid & 1;
    int row  = bid >> 1;
    int h    = row % Hh;
    int n    = row / Hh;
    int w0   = wseg * 64;
    float scale = fmaxf(__uint_as_float(slots[0]), 1e-12f) / QMAX;
    int t  = threadIdx.x;
    int wl = t & 63;
    int c0 = t >> 6;
    for (int it = 0; it < 64; ++it) {
        int c = it * 4 + c0;
        float v = x[(((size_t)n * Cc + c) * Hh + h) * Ww + w0 + wl];
        float r = rintf(v / scale);
        r = fminf(fmaxf(r, -128.0f), 127.0f);
        q[wl][c] = (char)(int)r;
    }
    __syncthreads();
    int wl2 = t >> 2;
    int cb  = (t & 3) * 64;
    const v4u* src = (const v4u*)&q[wl2][cb];
    v4u* dst = (v4u*)(xq + ((((size_t)n * Hh + h) * Ww + w0 + wl2) * Cc + cb));
    dst[0] = src[0]; dst[1] = src[1]; dst[2] = src[2]; dst[3] = src[3];
}

__global__ __launch_bounds__(256) void quant_w_kernel(const float* __restrict__ w,
                                                      const unsigned int* __restrict__ slots,
                                                      char* __restrict__ wq) {
    size_t idx = (size_t)blockIdx.x * 256 + threadIdx.x;
    if (idx >= W_ELEMS) return;
    int pos = (int)(idx / ((size_t)OC * Cc));
    int rem = (int)(idx % ((size_t)OC * Cc));
    int oc  = rem >> 8;
    int c   = rem & 255;
    int ch  = oc >> 7;
    float scale = fmaxf(__uint_as_float(slots[1 + ch]), 1e-12f) / QMAX;
    float v = w[((size_t)oc * Cc + c) * 9 + pos];
    float r = fminf(fmaxf(rintf(v / scale), -128.0f), 127.0f);
    wq[idx] = (char)(int)r;
}

__global__ __launch_bounds__(256) void conv_wmma_kernel(const char* __restrict__ xq,
                                                        const char* __restrict__ wq,
                                                        const float* __restrict__ bias,
                                                        const unsigned int* __restrict__ slots,
                                                        float* __restrict__ out) {
    __shared__ char wlds[2][COC * Cc];
    int bid   = blockIdx.x;
    int chunk = bid & 3;
    int row   = bid >> 2;
    int h     = row % Hh;
    int n     = row / Hh;
    int oc_base = chunk * COC;
    int lane = threadIdx.x & 31;
    int wave = threadIdx.x >> 5;
    int p    = wave >> 1;
    int q    = wave & 1;
    int r    = lane & 15;
    int half = lane >> 4;
    int wp   = p * 32;

    v8i acc[2][4];
#pragma unroll
    for (int m = 0; m < 2; ++m)
#pragma unroll
        for (int t = 0; t < 4; ++t) acc[m][t] = (v8i)0;

    const char* wq_chunk = wq + (size_t)oc_base * Cc;

    auto stage = [&](int pos, int buf) {
#if USE_ASYNC_LDS
        unsigned long long g = (unsigned long long)(size_t)(wq_chunk + (size_t)pos * OC * Cc);
        unsigned lbase = (unsigned)(size_t)&wlds[buf][0] + threadIdx.x * 128u;
        unsigned gbase = threadIdx.x * 128u;
#pragma unroll
        for (int i = 0; i < 8; ++i) {
            unsigned ldst = lbase + i * 16u;
            unsigned goff = gbase + i * 16u;
            asm volatile("global_load_async_to_lds_b128 %0, %1, %2"
                         :: "v"(ldst), "v"(goff), "s"(g) : "memory");
        }
#else
        const v4u* src = (const v4u*)(wq_chunk + (size_t)pos * OC * Cc) + threadIdx.x * 8;
        v4u* dst = (v4u*)&wlds[buf][0] + threadIdx.x * 8;
#pragma unroll
        for (int i = 0; i < 8; ++i) dst[i] = src[i];
#endif
    };

    stage(0, 0);
    for (int pos = 0; pos < 9; ++pos) {
        if (pos < 8) stage(pos + 1, (pos + 1) & 1);
#if USE_ASYNC_LDS
        if (pos < 8) asm volatile("s_wait_asynccnt 8" ::: "memory");
        else         asm volatile("s_wait_asynccnt 0" ::: "memory");
#endif
        __syncthreads();
        const char* wb = &wlds[pos & 1][0];

        int dy = pos / 3 - 1, dx = pos % 3 - 1;
        int hh = h + dy;
        bool vh = (unsigned)hh < (unsigned)Hh;
        const char* arow = xq + ((size_t)n * Hh + hh) * Ww * Cc;
        int ww0 = wp + r + dx;
        int ww1 = ww0 + 16;
        bool v0 = vh && ((unsigned)ww0 < (unsigned)Ww);
        bool v1 = vh && ((unsigned)ww1 < (unsigned)Ww);
        const char* ab0 = arow + (size_t)ww0 * Cc;
        const char* ab1 = arow + (size_t)ww1 * Cc;

#pragma unroll
        for (int ks = 0; ks < 4; ++ks) {
            v8i a0, a1;
#pragma unroll
            for (int pp = 0; pp < 4; ++pp) {
                int koff = ks * 64 + (pp >> 1) * 32 + half * 16 + (pp & 1) * 8;
                v2u d0 = v0 ? *(const v2u*)(ab0 + koff) : (v2u)0u;
                v2u d1 = v1 ? *(const v2u*)(ab1 + koff) : (v2u)0u;
                a0[2 * pp] = (int)d0.x; a0[2 * pp + 1] = (int)d0.y;
                a1[2 * pp] = (int)d1.x; a1[2 * pp + 1] = (int)d1.y;
            }
#pragma unroll
            for (int t = 0; t < 4; ++t) {
                const v4u* bp = (const v4u*)(wb + (((q * 64 + t * 16 + r) * Cc) + ks * 64 + half * 16));
                v4u b0v = bp[0];
                v4u b1v = bp[2];
                v8i b;
                b[0] = (int)b0v.x; b[1] = (int)b0v.y; b[2] = (int)b0v.z; b[3] = (int)b0v.w;
                b[4] = (int)b1v.x; b[5] = (int)b1v.y; b[6] = (int)b1v.z; b[7] = (int)b1v.w;
                acc[0][t] = __builtin_amdgcn_wmma_i32_16x16x64_iu8(
                    true, a0, true, b, acc[0][t], false, false);
                acc[1][t] = __builtin_amdgcn_wmma_i32_16x16x64_iu8(
                    true, a1, true, b, acc[1][t], false, false);
            }
        }
        __syncthreads();
    }

    float sx = fmaxf(__uint_as_float(slots[0]), 1e-12f) / QMAX;
    float sw = fmaxf(__uint_as_float(slots[1 + chunk]), 1e-12f) / QMAX;
    float s  = sx * sw;
#pragma unroll
    for (int m = 0; m < 2; ++m) {
        int wout = wp + m * 16 + half * 8;
#pragma unroll
        for (int t = 0; t < 4; ++t) {
            int oc = oc_base + q * 64 + t * 16 + r;
            float bv = bias[oc];
            float* op = out + ((((size_t)n * OC + oc) * Hh + h) * Ww + wout);
            v4f f0, f1;
#pragma unroll
            for (int j = 0; j < 4; ++j) {
                f0[j] = (float)acc[m][t][j]     * s + bv;
                f1[j] = (float)acc[m][t][4 + j] * s + bv;
            }
            *(v4f*)op       = f0;
            *((v4f*)op + 1) = f1;
        }
    }
}

extern "C" void kernel_launch(void* const* d_in, const int* in_sizes, int n_in,
                              void* d_out, int out_size, void* d_ws, size_t ws_size,
                              hipStream_t stream) {
    const float* x    = (const float*)d_in[0];
    const float* wt   = (const float*)d_in[1];
    const float* bias = (const float*)d_in[2];
    float* out        = (float*)d_out;

    unsigned int* slots = (unsigned int*)d_ws;
    char* xq = (char*)d_ws + WS_XQ_OFF;
    char* wq = (char*)d_ws + WS_WQ_OFF;

    zero_slots_kernel<<<1, 64, 0, stream>>>(slots);
    amax_x_kernel<<<2048, 256, 0, stream>>>(x, slots);
    amax_w_kernel<<<256, 256, 0, stream>>>(wt, slots);
    quant_x_kernel<<<Nn * Hh * 2, 256, 0, stream>>>(x, slots, xq);
    quant_w_kernel<<<(int)((W_ELEMS + 255) / 256), 256, 0, stream>>>(wt, slots, wq);
    conv_wmma_kernel<<<Nn * Hh * 4, 256, 0, stream>>>(xq, wq, bias, slots, out);
}
